// PCLSTM_29102698398076
// MI455X (gfx1250) — compile-verified, matrix operand order checked by reading, not run on hardware
//
#include <hip/hip_runtime.h>
#include <hip/hip_bf16.h>

#define B_       20
#define C_       395
#define H_       10
#define T_       8192
#define GATE_IN_ 800
#define NG_      40       // 4*H
#define KSTEPS   13       // ceil(395/32) per pass; 12 full + 1 tail
#define NTILES   3        // ceil(40/16)

typedef __attribute__((ext_vector_type(16))) __bf16 v16bf;
typedef __attribute__((ext_vector_type(8)))  float  v8f;

// K offset within a 32-wide K-step for 16-bit A fragments:
// lane half (0/1) holds K halves; element v -> K = v + 8*half + (v>=8 ? 8 : 0)
__device__ __forceinline__ int kofs16(int v, int half) {
    return v + 8 * half + ((v & 8) ? 8 : 0);
}

// Build a 16x32 bf16 A fragment from f32 source rows (stride T_ between K's).
// TAIL: clamp+zero elements with k >= C_.  lmask: per-lane value mask (y t==0).
template <bool TAIL, bool USE_LMASK>
__device__ __forceinline__ v16bf build_afrag(const float* __restrict__ src,
                                             int kbase, int half, float lmask) {
    v16bf a;
    #pragma unroll
    for (int v = 0; v < 16; ++v) {
        const int k  = kbase + kofs16(v, half);
        const int kc = TAIL ? ((k < C_) ? k : (C_ - 1)) : k;
        float val = src[(size_t)kc * T_];
        if (TAIL) val = (k < C_) ? val : 0.0f;
        if (USE_LMASK) val *= lmask;
        a[v] = (__bf16)val;
    }
    return a;
}

// ---------------------------------------------------------------------------
// Kernel 1: pre GEMM  (M = T*B rows, K = 395 x-pass + 395 y-pass, N = 40)
// via v_wmma_f32_16x16x32_bf16.  Output layout: pre[b][t][o*4+g] (f32).
// ---------------------------------------------------------------------------
__global__ __launch_bounds__(256) void pclstm_pre_gemm(
    const float* __restrict__ x, const float* __restrict__ y,
    const float* __restrict__ Wf, const float* __restrict__ bf,
    const float* __restrict__ Wi, const float* __restrict__ bi,
    const float* __restrict__ Wu, const float* __restrict__ bu,
    const float* __restrict__ Wo, const float* __restrict__ bo,
    float* __restrict__ pre)
{
    __shared__ __bf16 ldsW[2 * KSTEPS * NTILES * 32 * 16];   // 78 KB

    const int tid  = threadIdx.x;
    const int lane = tid & 31;
    const int wave = tid >> 5;
    const int half = lane >> 4;
    const int m    = lane & 15;

    // ---- stage swizzled B fragments: [pass][ks][nt][lane][v] ----
    for (int i = tid; i < 2 * KSTEPS * NTILES * 512; i += blockDim.x) {
        const int v  = i & 15;
        const int ln = (i >> 4) & 31;
        const int nt = (i >> 9) % NTILES;
        const int pk = i / (NTILES * 512);
        const int ks = pk % KSTEPS;
        const int p  = pk / KSTEPS;
        const int n  = nt * 16 + (ln & 15);
        const int kl = ks * 32 + kofs16(v, ln >> 4);
        float val = 0.0f;
        if (n < NG_ && kl < C_) {
            const int g = n / 10, o = n % 10;
            const float* Wg = (g == 0) ? Wf : (g == 1) ? Wi : (g == 2) ? Wu : Wo;
            const int col = p ? (kl + C_ + H_) : kl;     // Wy cols start at C+H
            val = Wg[o * GATE_IN_ + col];
        }
        ldsW[i] = (__bf16)val;
    }
    __syncthreads();

    // ---- per-lane bias + permuted output column ----
    float bias_n[NTILES];
    int   ocol_n[NTILES];
    #pragma unroll
    for (int nt = 0; nt < NTILES; ++nt) {
        const int n = nt * 16 + (lane & 15);
        float bv = 0.0f;
        int   oc = 0;
        if (n < NG_) {
            const int g = n / 10, o = n % 10;
            const float* bg = (g == 0) ? bf : (g == 1) ? bi : (g == 2) ? bu : bo;
            bv = bg[o];
            oc = o * 4 + g;                              // [o][g] permuted layout
        }
        bias_n[nt] = bv;
        ocol_n[nt] = oc;
    }

    #pragma unroll 1
    for (int it = 0; it < 2; ++it) {
        const int mt   = (blockIdx.x * 8 + wave) * 2 + it;   // 10240 tiles
        const int row0 = mt * 16;
        const int bb   = row0 / T_;
        const int t0   = row0 % T_;
        const int tm   = t0 + m;
        const float* srcx = x + (size_t)bb * C_ * T_ + tm;
        const int   tm1   = (tm > 0) ? (tm - 1) : 0;
        const float ymask = (tm > 0) ? 1.0f : 0.0f;
        const float* srcy = y + (size_t)bb * C_ * T_ + tm1;

        v8f acc[NTILES];
        #pragma unroll
        for (int nt = 0; nt < NTILES; ++nt)
            #pragma unroll
            for (int r = 0; r < 8; ++r) acc[nt][r] = bias_n[nt];

        // ---- x pass: 12 unconditional K-steps + masked tail ----
        #pragma unroll 1
        for (int ks = 0; ks < KSTEPS - 1; ++ks) {
            const v16bf a = build_afrag<false, false>(srcx, ks * 32, half, 1.0f);
            #pragma unroll
            for (int nt = 0; nt < NTILES; ++nt) {
                const v16bf bfrag = *(const v16bf*)
                    &ldsW[((ks * NTILES + nt) * 32 + lane) * 16];
                acc[nt] = __builtin_amdgcn_wmma_f32_16x16x32_bf16(
                    false, a, false, bfrag, (short)0, acc[nt], false, false);
            }
        }
        {
            const v16bf a = build_afrag<true, false>(srcx, (KSTEPS - 1) * 32, half, 1.0f);
            #pragma unroll
            for (int nt = 0; nt < NTILES; ++nt) {
                const v16bf bfrag = *(const v16bf*)
                    &ldsW[(((KSTEPS - 1) * NTILES + nt) * 32 + lane) * 16];
                acc[nt] = __builtin_amdgcn_wmma_f32_16x16x32_bf16(
                    false, a, false, bfrag, (short)0, acc[nt], false, false);
            }
        }
        // ---- y pass ----
        #pragma unroll 1
        for (int ks = 0; ks < KSTEPS - 1; ++ks) {
            const v16bf a = build_afrag<false, true>(srcy, ks * 32, half, ymask);
            #pragma unroll
            for (int nt = 0; nt < NTILES; ++nt) {
                const v16bf bfrag = *(const v16bf*)
                    &ldsW[(((KSTEPS + ks) * NTILES + nt) * 32 + lane) * 16];
                acc[nt] = __builtin_amdgcn_wmma_f32_16x16x32_bf16(
                    false, a, false, bfrag, (short)0, acc[nt], false, false);
            }
        }
        {
            const v16bf a = build_afrag<true, true>(srcy, (KSTEPS - 1) * 32, half, ymask);
            #pragma unroll
            for (int nt = 0; nt < NTILES; ++nt) {
                const v16bf bfrag = *(const v16bf*)
                    &ldsW[(((2 * KSTEPS - 1) * NTILES + nt) * 32 + lane) * 16];
                acc[nt] = __builtin_amdgcn_wmma_f32_16x16x32_bf16(
                    false, a, false, bfrag, (short)0, acc[nt], false, false);
            }
        }

        // D layout: vgpr r -> row (r + 8*half); column permuted to [o][g]
        #pragma unroll
        for (int nt = 0; nt < NTILES; ++nt) {
            const int n = nt * 16 + (lane & 15);
            if (n < NG_) {
                float* dst = pre + ((size_t)bb * T_ + t0 + 8 * half) * NG_ + ocol_n[nt];
                #pragma unroll
                for (int r = 0; r < 8; ++r) dst[(size_t)r * NG_] = acc[nt][r];
            }
        }
    }
}

// ---------------------------------------------------------------------------
// Kernel 2: sequential LSTM scan.  One wave per batch; lane = cell.
// pre row layout [o*4+g] -> one b128 load per lane per step.
// ---------------------------------------------------------------------------
__device__ __forceinline__ float sigmoid_f(float z) {
    return __frcp_rn(1.0f + __expf(-z));
}
__device__ __forceinline__ float tanh_f(float z) {
    const float e = __expf(-2.0f * z);
    return (1.0f - e) * __frcp_rn(1.0f + e);
}

__global__ __launch_bounds__(32) void pclstm_scan(
    const float* __restrict__ pre,
    const float* __restrict__ Wf, const float* __restrict__ Wi,
    const float* __restrict__ Wu, const float* __restrict__ Wo,
    const float* __restrict__ b_init,
    float* __restrict__ out, float* __restrict__ cT)
{
    const int b    = blockIdx.x;
    const int lane = threadIdx.x;
    const bool act = lane < H_;
    const int  oo  = act ? lane : 0;

    float wf[H_], wi[H_], wu[H_], wo[H_];
    #pragma unroll
    for (int h = 0; h < H_; ++h) {
        wf[h] = Wf[oo * GATE_IN_ + C_ + h];
        wi[h] = Wi[oo * GATE_IN_ + C_ + h];
        wu[h] = Wu[oo * GATE_IN_ + C_ + h];
        wo[h] = Wo[oo * GATE_IN_ + C_ + h];
    }
    float c    = b_init[oo];       // zeros @ W_init.T + b_init == b_init
    float hcur = 0.0f;

    __shared__ float buf[H_][33];
    const float* pb = pre + (size_t)b * T_ * NG_;
    float*       ob = out + (size_t)b * H_ * T_;

    for (int t = 0; t < T_; ++t) {
        const float* p = pb + (size_t)t * NG_;
        if ((t & 15) == 0)
            __builtin_prefetch(p + NG_ * 16 + lane * 16, 0, 1);

        const float4 q = *(const float4*)(p + oo * 4);   // [f,i,u,o] pre-acts
        float z0 = q.x, z1 = q.y, z2 = q.z, z3 = q.w;
        #pragma unroll
        for (int h = 0; h < H_; ++h) {
            const float hv = __shfl(hcur, h, 32);
            z0 = fmaf(wf[h], hv, z0);
            z1 = fmaf(wi[h], hv, z1);
            z2 = fmaf(wu[h], hv, z2);
            z3 = fmaf(wo[h], hv, z3);
        }
        const float fg = sigmoid_f(z0);
        const float ig = sigmoid_f(z1);
        const float gg = tanh_f(z2);
        const float og = sigmoid_f(z3);
        c    = fmaf(c, fg, ig * gg);
        hcur = og * tanh_f(c);

        if (act) buf[oo][t & 31] = hcur;
        if ((t & 31) == 31) {
            __syncthreads();
            const int tb = t - 31;
            #pragma unroll
            for (int r = 0; r < H_; ++r)
                ob[(size_t)r * T_ + tb + lane] = buf[r][lane];
            __syncthreads();
        }
    }
    if (act) cT[b * H_ + lane] = c;
}

extern "C" void kernel_launch(void* const* d_in, const int* in_sizes, int n_in,
                              void* d_out, int out_size, void* d_ws, size_t ws_size,
                              hipStream_t stream) {
    const float* x  = (const float*)d_in[0];
    const float* y  = (const float*)d_in[1];
    const float* Wf = (const float*)d_in[2];
    const float* bf = (const float*)d_in[3];
    const float* Wi = (const float*)d_in[4];
    const float* bi = (const float*)d_in[5];
    const float* Wu = (const float*)d_in[6];
    const float* bu = (const float*)d_in[7];
    const float* Wo = (const float*)d_in[8];
    const float* bo = (const float*)d_in[9];
    // d_in[10] = W_init (unused: zeros @ W_init.T == 0)
    const float* b_init = (const float*)d_in[11];

    float* pre = (float*)d_ws;                      // [B][T][40] f32, 26.2 MB
    float* out = (float*)d_out;                     // (B,H,T)
    float* cT  = out + (size_t)B_ * H_ * T_;        // (B,H)

    pclstm_pre_gemm<<<640, 256, 0, stream>>>(x, y, Wf, bf, Wi, bi, Wu, bu, Wo, bo, pre);
    pclstm_scan<<<B_, 32, 0, stream>>>(pre, Wf, Wi, Wu, Wo, b_init, out, cT);
}
